// MultiHeadAttention_21741124452899
// MI455X (gfx1250) — compile-verified, matrix operand order checked by reading, not run on hardware
//
#include <hip/hip_runtime.h>
#include <cstdint>

// Problem constants (reference: B=2, S=2048, D=1024, H=16, DK=64)
#define BB 2
#define SS 2048
#define DD 1024
#define HH 16
#define DKK 64
#define GM 4096   // BB*SS
#define GN 1024
#define GK 1024

typedef __attribute__((ext_vector_type(16))) __bf16 v16bf;
typedef __attribute__((ext_vector_type(8)))  __bf16 v8bf;
typedef __attribute__((ext_vector_type(8)))  float  v8f;
typedef __attribute__((ext_vector_type(4)))  float  v4f;

// ---- CDNA5 async global->LDS copy (ASYNCcnt path) ------------------------
__device__ __forceinline__ uint32_t lds_addr32(const void* p) {
    return (uint32_t)(uintptr_t)p;   // low 32 bits of LDS-aperture address
}
__device__ __forceinline__ void async_b128(uint32_t lds_off, const void* g) {
    asm volatile("global_load_async_to_lds_b128 %0, %1, off"
                 :: "v"(lds_off), "v"((uint64_t)(uintptr_t)g)
                 : "memory");
}
__device__ __forceinline__ void wait_async0() {
    asm volatile("s_wait_asynccnt 0x0" ::: "memory");
}

// ---- WMMA helpers ---------------------------------------------------------
__device__ __forceinline__ v16bf frag_ld(const __bf16* p0, const __bf16* p1) {
    v8bf lo = *reinterpret_cast<const v8bf*>(p0);
    v8bf hi = *reinterpret_cast<const v8bf*>(p1);
    return __builtin_shufflevector(lo, hi, 0,1,2,3,4,5,6,7,8,9,10,11,12,13,14,15);
}
__device__ __forceinline__ v8f wmma_bf16(v16bf a, v16bf b, v8f c) {
    return __builtin_amdgcn_wmma_f32_16x16x32_bf16(false, a, false, b, (short)0, c,
                                                   false, false);
}

// ---------------------------------------------------------------------------
// One-shot converters
// ---------------------------------------------------------------------------
__global__ __launch_bounds__(256)
void pack_bf16(const float* __restrict__ in, __bf16* __restrict__ out) {
    const int i = (blockIdx.x * 256 + threadIdx.x) * 8;
    v4f a = *reinterpret_cast<const v4f*>(in + i);
    v4f b = *reinterpret_cast<const v4f*>(in + i + 4);
    v8bf r;
#pragma unroll
    for (int j = 0; j < 4; ++j) { r[j] = (__bf16)a[j]; r[4 + j] = (__bf16)b[j]; }
    *reinterpret_cast<v8bf*>(out + i) = r;
}

// W[k][n] fp32 -> Wt[n][k] bf16
__global__ __launch_bounds__(256)
void transpose_pack_bf16(const float* __restrict__ W, __bf16* __restrict__ Wt) {
    __shared__ __bf16 tile[64][72];
    const int t  = threadIdx.x;
    const int k0 = blockIdx.y * 64, n0 = blockIdx.x * 64;
    {
        const int k = t >> 2, nb = (t & 3) * 16;
        const float* src = W + (size_t)(k0 + k) * GN + n0 + nb;
#pragma unroll
        for (int q = 0; q < 4; ++q) {
            v4f f = *reinterpret_cast<const v4f*>(src + q * 4);
#pragma unroll
            for (int j = 0; j < 4; ++j) tile[nb + q * 4 + j][k] = (__bf16)f[j];
        }
    }
    __syncthreads();
    {
        const int n = t >> 2, kb = (t & 3) * 16;
        v8bf a = *reinterpret_cast<v8bf*>(&tile[n][kb]);
        v8bf b = *reinterpret_cast<v8bf*>(&tile[n][kb + 8]);
        __bf16* dst = Wt + (size_t)(n0 + n) * GK + k0 + kb;
        *reinterpret_cast<v8bf*>(dst)     = a;
        *reinterpret_cast<v8bf*>(dst + 8) = b;
    }
}

// ---------------------------------------------------------------------------
// GEMM: Y[M,N] = A[M,K](bf16) @ Bt[N,K]^T(bf16) + bias(f32).
// Tile 256x64xK64, double-buffered async staging (copy overlaps WMMA).
// mode 0: f32 [M,N]; mode 1: bf16 [M,N] scaled; mode 2: bf16 Vt[b][h][dk][s].
// ---------------------------------------------------------------------------
#define ABUF (256 * 144)   // bytes per A buffer
#define BBUF (64 * 144)

__global__ __launch_bounds__(256)
void gemm_bf16(const __bf16* __restrict__ A, const __bf16* __restrict__ Bt,
               const float* __restrict__ bias, void* __restrict__ Y,
               int mode, float scale) {
    __shared__ __bf16 Alds[2][256][72];
    __shared__ __bf16 Blds[2][64][72];

    const int tid   = threadIdx.x;
    const int wave  = tid >> 5;
    const int lane  = tid & 31;
    const int l16   = lane & 15;
    const int lhalf = lane >> 4;
    const int m0    = blockIdx.y * 256;
    const int n0    = blockIdx.x * 64;

    // Per-thread staging geometry: chunk i = tid&7 fixed, row = tid>>3 + 32*j.
    const int srow = tid >> 3, si = tid & 7;
    const uint32_t aOff = lds_addr32(&Alds[0][0][0]) + srow * 144 + si * 16;
    const uint32_t bOff = lds_addr32(&Blds[0][0][0]) + srow * 144 + si * 16;
    const __bf16* aG = A  + (size_t)(m0 + srow) * GK + si * 8;
    const __bf16* bG = Bt + (size_t)(n0 + srow) * GK + si * 8;

    const v8f zero = {0.f, 0.f, 0.f, 0.f, 0.f, 0.f, 0.f, 0.f};
    v8f acc[2][4] = {{zero, zero, zero, zero}, {zero, zero, zero, zero}};

    auto stage = [&](int buf, int k0) {
#pragma unroll
        for (int j = 0; j < 8; ++j)   // A tile: 256 rows x 64 K
            async_b128(aOff + buf * ABUF + j * (32 * 144),
                       aG + k0 + (size_t)j * 32 * GK);
#pragma unroll
        for (int j = 0; j < 2; ++j)   // B tile: 64 rows x 64 K
            async_b128(bOff + buf * BBUF + j * (32 * 144),
                       bG + k0 + (size_t)j * 32 * GK);
    };

    stage(0, 0);
    const int iters = GK / 64;
    for (int it = 0; it < iters; ++it) {
        const int buf = it & 1;
        wait_async0();
        __syncthreads();
        if (it + 1 < iters) stage(buf ^ 1, (it + 1) * 64);

#pragma unroll
        for (int kk = 0; kk < 2; ++kk) {
            const int ko = kk * 32 + lhalf * 8;
            v16bf a0 = frag_ld(&Alds[buf][wave * 32 + l16][ko],
                               &Alds[buf][wave * 32 + l16][ko + 16]);
            v16bf a1 = frag_ld(&Alds[buf][wave * 32 + 16 + l16][ko],
                               &Alds[buf][wave * 32 + 16 + l16][ko + 16]);
#pragma unroll
            for (int nt = 0; nt < 4; ++nt) {
                v16bf bfr = frag_ld(&Blds[buf][nt * 16 + l16][ko],
                                    &Blds[buf][nt * 16 + l16][ko + 16]);
                acc[0][nt] = wmma_bf16(a0, bfr, acc[0][nt]);
                acc[1][nt] = wmma_bf16(a1, bfr, acc[1][nt]);
            }
        }
        __syncthreads();
    }

#pragma unroll
    for (int mm = 0; mm < 2; ++mm) {
#pragma unroll
        for (int nt = 0; nt < 4; ++nt) {
            const int col = n0 + nt * 16 + l16;
            const float bv = bias[col];
#pragma unroll
            for (int r = 0; r < 8; ++r) {
                const int row = m0 + wave * 32 + mm * 16 + lhalf * 8 + r;
                const float val = (acc[mm][nt][r] + bv) * scale;
                if (mode == 0) {
                    ((float*)Y)[(size_t)row * GN + col] = val;
                } else if (mode == 1) {
                    ((__bf16*)Y)[(size_t)row * GN + col] = (__bf16)val;
                } else {  // Vt[b][h][dk][s]
                    const int b = row >> 11, s = row & (SS - 1);
                    const int h = col >> 6,  dk = col & (DKK - 1);
                    ((__bf16*)Y)[(((size_t)b * HH + h) * DKK + dk) * SS + s] =
                        (__bf16)val;
                }
            }
        }
    }
}

// ---------------------------------------------------------------------------
// Flash attention. Qp bf16 [B,S,D] (pre-scaled 1/sqrt(DK)), Kp bf16 [B,S,D],
// Vt bf16 [B,H,DK,S], O bf16 [B,S,D]. Workgroup = (b,h,128 q-rows); 8 waves x
// 16 rows. 64-key chunks, double-buffered async staging.
// ---------------------------------------------------------------------------
#define KVBUF (64 * 144)   // bytes per K or V buffer

__global__ __launch_bounds__(256)
void mha_flash_attn(const __bf16* __restrict__ Qp, const __bf16* __restrict__ Kp,
                    const __bf16* __restrict__ Vt, __bf16* __restrict__ O) {
    __shared__ __bf16 Klds[2][64][72];     // [key][dk]
    __shared__ __bf16 Vlds[2][64][72];     // [dk][key]
    __shared__ __bf16 Plds[8][16][72];     // per-wave P scratch [row][key]

    const int tid   = threadIdx.x;
    const int wave  = tid >> 5;
    const int lane  = tid & 31;
    const int l16   = lane & 15;
    const int lhalf = lane >> 4;
    const int b     = blockIdx.y / HH;
    const int h     = blockIdx.y % HH;
    const int qr0   = blockIdx.x * 128 + wave * 16;
    const int kb    = lhalf * 8;

    // Q fragments (dk = 64 -> two K=32 fragments)
    const __bf16* qptr = Qp + ((size_t)b * SS + qr0 + l16) * DD + h * DKK;
    v16bf aq[2];
#pragma unroll
    for (int c = 0; c < 2; ++c) {
        const int ko = c * 32 + kb;
        aq[c] = frag_ld(qptr + ko, qptr + ko + 16);
    }

    // Staging geometry: chunk i = tid&7 fixed, row = tid>>3 + 32*j (j=0,1)
    const int srow = tid >> 3, si = tid & 7;
    const uint32_t kOff = lds_addr32(&Klds[0][0][0]) + srow * 144 + si * 16;
    const uint32_t vOff = lds_addr32(&Vlds[0][0][0]) + srow * 144 + si * 16;
    const __bf16* kG = Kp + ((size_t)b * SS + srow) * DD + h * DKK + si * 8;
    const __bf16* vG = Vt + (((size_t)b * HH + h) * DKK + srow) * SS + si * 8;

    auto stage = [&](int buf, int kv) {
#pragma unroll
        for (int j = 0; j < 2; ++j) {     // K tile: 64 keys x 64 dk
            async_b128(kOff + buf * KVBUF + j * (32 * 144),
                       kG + ((size_t)kv + j * 32) * DD);
            async_b128(vOff + buf * KVBUF + j * (32 * 144),
                       vG + (size_t)j * 32 * SS + kv);   // V tile: 64 dk x 64 keys
        }
    };

    const v8f zero = {0.f, 0.f, 0.f, 0.f, 0.f, 0.f, 0.f, 0.f};
    v8f   o_acc[4] = {zero, zero, zero, zero};
    float mrow[8], lrow[8];
#pragma unroll
    for (int r = 0; r < 8; ++r) { mrow[r] = -3.0e38f; lrow[r] = 0.f; }

    stage(0, 0);
    const int chunks = SS / 64;
    for (int it = 0; it < chunks; ++it) {
        const int buf = it & 1;
        wait_async0();
        __syncthreads();
        if (it + 1 < chunks) stage(buf ^ 1, (it + 1) * 64);

        // scores S[16 x 64] : 8 WMMAs (key tiles nt, dk chunks c)
        v8f s[4] = {zero, zero, zero, zero};
#pragma unroll
        for (int c = 0; c < 2; ++c) {
            const int ko = c * 32 + kb;
#pragma unroll
            for (int nt = 0; nt < 4; ++nt) {
                v16bf bk = frag_ld(&Klds[buf][nt * 16 + l16][ko],
                                   &Klds[buf][nt * 16 + l16][ko + 16]);
                s[nt] = wmma_bf16(aq[c], bk, s[nt]);
            }
        }

        // online softmax over 64 keys
#pragma unroll
        for (int r = 0; r < 8; ++r) {
            float cm = fmaxf(fmaxf(s[0][r], s[1][r]), fmaxf(s[2][r], s[3][r]));
#pragma unroll
            for (int off = 8; off > 0; off >>= 1)
                cm = fmaxf(cm, __shfl_xor(cm, off, 32));
            const float mnew = fmaxf(mrow[r], cm);
            const float corr = __expf(mrow[r] - mnew);
            float p[4], rs = 0.f;
#pragma unroll
            for (int j = 0; j < 4; ++j) { p[j] = __expf(s[j][r] - mnew); rs += p[j]; }
#pragma unroll
            for (int off = 8; off > 0; off >>= 1)
                rs += __shfl_xor(rs, off, 32);
            lrow[r] = lrow[r] * corr + rs;
            mrow[r] = mnew;
#pragma unroll
            for (int nt = 0; nt < 4; ++nt) o_acc[nt][r] *= corr;
            const int prow = lhalf * 8 + r;
#pragma unroll
            for (int j = 0; j < 4; ++j)
                Plds[wave][prow][j * 16 + l16] = (__bf16)p[j];
        }

        // O += P[16x64] @ V[64x64] : 8 WMMAs
        v16bf ap0 = frag_ld(&Plds[wave][l16][kb],      &Plds[wave][l16][kb + 16]);
        v16bf ap1 = frag_ld(&Plds[wave][l16][32 + kb], &Plds[wave][l16][32 + kb + 16]);
#pragma unroll
        for (int nt = 0; nt < 4; ++nt) {
            v16bf bv0 = frag_ld(&Vlds[buf][nt * 16 + l16][kb],
                                &Vlds[buf][nt * 16 + l16][kb + 16]);
            v16bf bv1 = frag_ld(&Vlds[buf][nt * 16 + l16][32 + kb],
                                &Vlds[buf][nt * 16 + l16][32 + kb + 16]);
            o_acc[nt] = wmma_bf16(ap0, bv0, o_acc[nt]);
            o_acc[nt] = wmma_bf16(ap1, bv1, o_acc[nt]);
        }
        __syncthreads();
    }

    // epilogue: normalize, store bf16 [B,S,D]
#pragma unroll
    for (int nt = 0; nt < 4; ++nt) {
        const int col = h * DKK + nt * 16 + l16;
#pragma unroll
        for (int r = 0; r < 8; ++r) {
            const int mr = lhalf * 8 + r;
            O[((size_t)b * SS + qr0 + mr) * DD + col] =
                (__bf16)(o_acc[nt][r] / lrow[r]);
        }
    }
}

// ---------------------------------------------------------------------------
extern "C" void kernel_launch(void* const* d_in, const int* in_sizes, int n_in,
                              void* d_out, int out_size, void* d_ws, size_t ws_size,
                              hipStream_t stream) {
    const float* q   = (const float*)d_in[0];
    const float* k   = (const float*)d_in[1];
    const float* v   = (const float*)d_in[2];
    const float* w_q = (const float*)d_in[3];
    const float* b_q = (const float*)d_in[4];
    const float* w_k = (const float*)d_in[5];
    const float* b_k = (const float*)d_in[6];
    const float* w_v = (const float*)d_in[7];
    const float* b_v = (const float*)d_in[8];
    const float* w_o = (const float*)d_in[9];
    const float* b_o = (const float*)d_in[10];
    float* out = (float*)d_out;

    char* ws = (char*)d_ws;
    const size_t actEl = (size_t)GM * GN;                   // 4M elements
    const size_t actB  = actEl * sizeof(__bf16);            // 8 MB
    const size_t wB    = (size_t)GK * GN * sizeof(__bf16);  // 2 MB
    __bf16* qb  = (__bf16*)(ws);
    __bf16* kbuf= (__bf16*)(ws + actB);
    __bf16* vb  = (__bf16*)(ws + 2 * actB);
    __bf16* wqt = (__bf16*)(ws + 3 * actB);
    __bf16* wkt = (__bf16*)(ws + 3 * actB + wB);
    __bf16* wvt = (__bf16*)(ws + 3 * actB + 2 * wB);
    __bf16* wot = (__bf16*)(ws + 3 * actB + 3 * wB);
    __bf16* qp  = (__bf16*)(ws + 3 * actB + 4 * wB);
    __bf16* kp  = (__bf16*)(ws + 4 * actB + 4 * wB);
    __bf16* vt  = (__bf16*)(ws + 5 * actB + 4 * wB);
    __bf16* op  = (__bf16*)(ws + 6 * actB + 4 * wB);

    dim3 blk(256);
    const int packBlocks = (int)(actEl / 2048);
    pack_bf16<<<packBlocks, blk, 0, stream>>>(q, qb);
    pack_bf16<<<packBlocks, blk, 0, stream>>>(k, kbuf);
    pack_bf16<<<packBlocks, blk, 0, stream>>>(v, vb);

    dim3 tgrid(GN / 64, GK / 64);
    transpose_pack_bf16<<<tgrid, blk, 0, stream>>>(w_q, wqt);
    transpose_pack_bf16<<<tgrid, blk, 0, stream>>>(w_k, wkt);
    transpose_pack_bf16<<<tgrid, blk, 0, stream>>>(w_v, wvt);
    transpose_pack_bf16<<<tgrid, blk, 0, stream>>>(w_o, wot);

    dim3 ggrid(GN / 64, GM / 256);
    const float qscale = 0.125f;   // 1/sqrt(DK)
    gemm_bf16<<<ggrid, blk, 0, stream>>>(qb,   wqt, b_q, qp,  1, qscale);
    gemm_bf16<<<ggrid, blk, 0, stream>>>(kbuf, wkt, b_k, kp,  1, 1.0f);
    gemm_bf16<<<ggrid, blk, 0, stream>>>(vb,   wvt, b_v, vt,  2, 1.0f);

    dim3 agrid(SS / 128, BB * HH);
    mha_flash_attn<<<agrid, blk, 0, stream>>>(qp, kp, vt, op);

    gemm_bf16<<<ggrid, blk, 0, stream>>>(op, wot, b_o, out, 0, 1.0f);
}
